// AttentionBlock_18502719111732
// MI455X (gfx1250) — matrix operand order repaired and checked by reading, not compiled, not run on hardware
//
#include <hip/hip_runtime.h>
#include <hip/hip_bf16.h>
#include <math.h>

// ---------------- problem constants ----------------
#define N_B   4
#define C_DIM 256
#define L_DIM 4096
#define AGGR  4
#define L2_DIM (L_DIM / AGGR)     // 1024
#define HEADS 4
#define E_DIM 64
#define BN_EPS 1e-5f

// ---------------- WMMA plumbing ----------------
typedef __attribute__((ext_vector_type(16))) __bf16 v16bf;
typedef __attribute__((ext_vector_type(8)))  __bf16 v8bf;
typedef __attribute__((ext_vector_type(8)))  float  v8f;

__device__ __forceinline__ v8bf lds_ld8(const __bf16* p) {
  return *(const v8bf*)p;                       // 16B-aligned -> ds_load_b128
}
__device__ __forceinline__ v16bf cat16(v8bf a, v8bf b) {
  return __builtin_shufflevector(a, b, 0,1,2,3,4,5,6,7,8,9,10,11,12,13,14,15);
}
__device__ __forceinline__ v8f wmma_bf16(v16bf a, v16bf b, v8f c) {
  return __builtin_amdgcn_wmma_f32_16x16x32_bf16(false, a, false, b, (short)0, c,
                                                 false, false);
}

// ---------------- async global->LDS copy (gfx1250), guarded fallback ----------------
#if defined(__gfx1250__) && __has_builtin(__builtin_amdgcn_global_load_async_to_lds_b128) && \
    __has_builtin(__builtin_amdgcn_s_wait_asynccnt)
#define USE_ASYNC 1
#else
#define USE_ASYNC 0
#endif

// pointee type of the async-copy builtins: GCC-vector int4 (per hipcc diagnostic)
typedef int v4i_g __attribute__((vector_size(16)));
typedef __attribute__((address_space(1))) v4i_g* g_v4i_ptr;
typedef __attribute__((address_space(3))) v4i_g* l_v4i_ptr;

__device__ __forceinline__ void cp16(void* lds_dst, const void* gsrc) {
#if USE_ASYNC
  __builtin_amdgcn_global_load_async_to_lds_b128(
      (g_v4i_ptr)gsrc, (l_v4i_ptr)lds_dst, 0, 0);
#else
  *reinterpret_cast<uint4*>(lds_dst) = *reinterpret_cast<const uint4*>(gsrc);
#endif
}
__device__ __forceinline__ void cp_wait() {
#if USE_ASYNC
  __builtin_amdgcn_s_wait_asynccnt(0);
#endif
}

// ---------------- kernel 1: fold BN1∘BN2 into per-channel scale/shift ----------------
__global__ void bn_coeffs_kernel(const float* __restrict__ g1, const float* __restrict__ b1,
                                 const float* __restrict__ m1, const float* __restrict__ v1,
                                 const float* __restrict__ g2, const float* __restrict__ b2,
                                 const float* __restrict__ m2, const float* __restrict__ v2,
                                 float* __restrict__ s, float* __restrict__ t) {
  int i = threadIdx.x;                          // 256 channels
  float i1 = g1[i] * rsqrtf(v1[i] + BN_EPS);
  float t1 = b1[i] - m1[i] * i1;
  float i2 = g2[i] * rsqrtf(v2[i] + BN_EPS);
  float t2 = b2[i] - m2[i] * i2;
  s[i] = i1 * i2;
  t[i] = t1 * i2 + t2;
}

// ---------------- kernel 2: avg+max pool over AGGR=4 ----------------
__global__ void pool_kernel(const float* __restrict__ x, float* __restrict__ xa) {
  size_t i = (size_t)blockIdx.x * blockDim.x + threadIdx.x;   // N*C*L2 outputs
  float4 p = reinterpret_cast<const float4*>(x)[i];
  float mx = fmaxf(fmaxf(p.x, p.y), fmaxf(p.z, p.w));
  xa[i] = (p.x + p.y + p.z + p.w) * 0.25f + mx;
}

// ---------------- kernel 3: generic 256x256 channel GEMM via WMMA ----------------
// Y[n, o, l] = ((sum_c W[o,c] * X[n,c,l]) * scale[o] + shift[o]) * cscale
// Block: 256 threads (8 waves). Tile: 128 (out-ch) x 64 (positions). K-step 32 (bf16).
// X_T / OUT_T : tensor stored position-major [Lx][C] instead of [C][Lx]
// X_BF16 / OUT_BF16 : tensor element type is bf16 instead of f32
#define GPA 40   // LDS pitch (bf16) for W tile, 80B (16B-aligned rows)
#define GPB 40   // LDS pitch (bf16) for X tile

template <bool X_T, bool OUT_T, bool HAS_SCALE, bool X_BF16, bool OUT_BF16>
__global__ __launch_bounds__(256)
void gemm256_kernel(const float* __restrict__ W, const void* __restrict__ Xv,
                    void* __restrict__ Yv, const float* __restrict__ scale,
                    const float* __restrict__ shift, float cscale, int Lx) {
  __shared__ __attribute__((aligned(16))) __bf16 Ws[128 * GPA];
  __shared__ __attribute__((aligned(16))) __bf16 Xs[64 * GPB];

  const int tid   = threadIdx.x;
  const int wave  = tid >> 5;
  const int lane  = tid & 31;
  const int laneN = lane & 15;
  const int khalf = lane >> 4;
  const int wrow0 = wave * 16;

  const int n0 = blockIdx.x * 64;          // position tile
  const int m0 = blockIdx.y * 128;         // out-channel tile
  const int n  = blockIdx.z;               // batch

  v8f acc[4];
#pragma unroll
  for (int i = 0; i < 4; ++i) acc[i] = (v8f){};

  for (int k0 = 0; k0 < C_DIM; k0 += 32) {
    // ---- stage X tile 32(k)x64(l) -> bf16 LDS [l][k] ----
    if constexpr (X_T && X_BF16) {
      // bf16 pos-major: pure 16B copies (async on gfx1250), 1 chunk/thread
      const __bf16* Xb = (const __bf16*)Xv + (size_t)n * C_DIM * Lx;
      int row = tid >> 2, q = tid & 3;
      cp16(Xs + row * GPB + q * 8, Xb + (size_t)(n0 + row) * C_DIM + k0 + q * 8);
    } else if constexpr (!X_T && !X_BF16) {
      // f32 channel-major: load along l, transpose+convert into LDS
      const float* Xb = (const float*)Xv + (size_t)n * C_DIM * Lx;
      int c = tid >> 3, lq = tid & 7;      // c local 0..31, 8 positions each
      const float* src = Xb + (size_t)(k0 + c) * Lx + n0 + lq * 8;
#pragma unroll
      for (int j = 0; j < 2; ++j) {
        float4 f = reinterpret_cast<const float4*>(src)[j];
        Xs[(lq * 8 + j * 4 + 0) * GPB + c] = (__bf16)f.x;
        Xs[(lq * 8 + j * 4 + 1) * GPB + c] = (__bf16)f.y;
        Xs[(lq * 8 + j * 4 + 2) * GPB + c] = (__bf16)f.z;
        Xs[(lq * 8 + j * 4 + 3) * GPB + c] = (__bf16)f.w;
      }
    }
    // ---- stage W tile 128x32 f32 -> bf16 LDS [m][k] (overlaps async X copy) ----
    {
      int row = tid >> 1, half = tid & 1;
      const float* src = W + (size_t)(m0 + row) * C_DIM + k0 + half * 16;
      __bf16* dst = Ws + row * GPA + half * 16;
#pragma unroll
      for (int j = 0; j < 4; ++j) {
        float4 f = reinterpret_cast<const float4*>(src)[j];
        dst[j * 4 + 0] = (__bf16)f.x; dst[j * 4 + 1] = (__bf16)f.y;
        dst[j * 4 + 2] = (__bf16)f.z; dst[j * 4 + 3] = (__bf16)f.w;
      }
    }
    if constexpr (X_T && X_BF16) cp_wait();
    __syncthreads();

    // ---- WMMA: A = W rows (16x32), B = X cols (32x16) x 4 ----
    const __bf16* arow = Ws + (wrow0 + laneN) * GPA;
    v16bf a = cat16(lds_ld8(arow + khalf * 16), lds_ld8(arow + khalf * 16 + 8));
#pragma unroll
    for (int nt = 0; nt < 4; ++nt) {
      const __bf16* bcol = Xs + (nt * 16 + laneN) * GPB + khalf * 16;
      v16bf b = cat16(lds_ld8(bcol), lds_ld8(bcol + 8));
      acc[nt] = wmma_bf16(a, b, acc[nt]);
    }
    __syncthreads();
  }

  // ---- epilogue: scale/shift + store (f32 or bf16) ----
#pragma unroll
  for (int r = 0; r < 8; ++r) {
    int o = m0 + wrow0 + r + 8 * khalf;
    float sc = HAS_SCALE ? scale[o] : 1.0f;
    float sh = shift[o];
#pragma unroll
    for (int nt = 0; nt < 4; ++nt) {
      int l = n0 + nt * 16 + laneN;
      float y = (acc[nt][r] * sc + sh) * cscale;
      size_t idx = OUT_T ? ((size_t)l * C_DIM + o) : ((size_t)o * Lx + l);
      if constexpr (OUT_BF16)
        ((__bf16*)Yv + (size_t)n * C_DIM * Lx)[idx] = (__bf16)y;
      else
        ((float*)Yv + (size_t)n * C_DIM * Lx)[idx] = y;
    }
  }
}

// ---------------- kernel 4: flash-style attention (bf16 in/out, f32 math) ----------------
// Q: [N][L][C] bf16 pos-major (softmax scale pre-folded), K: [N][L2][C] bf16,
// V: [N][C][L2] bf16, O: [N][L][C] bf16
#define APQ 72   // LDS pitch (bf16), 144B rows (16B-aligned)

__global__ __launch_bounds__(128)
void attn_kernel(const __bf16* __restrict__ Qb, const __bf16* __restrict__ Kb,
                 const __bf16* __restrict__ Vb, __bf16* __restrict__ Ob) {
  __shared__ __attribute__((aligned(16))) __bf16 Qs[64 * APQ];
  __shared__ __attribute__((aligned(16))) __bf16 Ks[64 * APQ];
  __shared__ __attribute__((aligned(16))) __bf16 Vs[64 * APQ];
  __shared__ __attribute__((aligned(16))) __bf16 Ps[64 * APQ];

  const int tid   = threadIdx.x;               // 128 threads, 4 waves
  const int wave  = tid >> 5;
  const int lane  = tid & 31;
  const int laneN = lane & 15;
  const int khalf = lane >> 4;
  const int wrow0 = wave * 16;

  const int l0 = blockIdx.x * 64;              // query tile
  const int h  = blockIdx.y;
  const int n  = blockIdx.z;

  // ---- stage Q tile (64 l x 64 e): pure byte copy, 4 x 16B per thread ----
#pragma unroll
  for (int i = 0; i < 4; ++i) {
    int chunk = i * 128 + tid;                 // 512 chunks of 16B
    int row = chunk >> 3, c16 = chunk & 7;
    cp16(Qs + row * APQ + c16 * 8,
         Qb + ((size_t)(n * L_DIM + l0 + row)) * C_DIM + h * E_DIM + c16 * 8);
  }

  float rmax[8], rsum[8];
  v8f acco[4];
#pragma unroll
  for (int r = 0; r < 8; ++r) { rmax[r] = -1.0e30f; rsum[r] = 0.0f; }
#pragma unroll
  for (int i = 0; i < 4; ++i) acco[i] = (v8f){};

  for (int kt = 0; kt < L2_DIM / 64; ++kt) {
    // ---- stage K tile (64 m x 64 e) and V tile (64 e x 64 m): byte copies ----
#pragma unroll
    for (int i = 0; i < 4; ++i) {
      int chunk = i * 128 + tid;
      int row = chunk >> 3, c16 = chunk & 7;
      cp16(Ks + row * APQ + c16 * 8,
           Kb + ((size_t)(n * L2_DIM + kt * 64 + row)) * C_DIM + h * E_DIM + c16 * 8);
      cp16(Vs + row * APQ + c16 * 8,
           Vb + ((size_t)(n * C_DIM + h * E_DIM + row)) * L2_DIM + kt * 64 + c16 * 8);
    }
    cp_wait();
    __syncthreads();

    // ---- S = q^T k : 64x64 per block, 16x64 per wave ----
    v8f s[4];
#pragma unroll
    for (int i = 0; i < 4; ++i) s[i] = (v8f){};
#pragma unroll
    for (int ks = 0; ks < 2; ++ks) {           // E = 64 -> two K-steps of 32
      const __bf16* qrow = Qs + (wrow0 + laneN) * APQ + ks * 32;
      v16bf a = cat16(lds_ld8(qrow + khalf * 16), lds_ld8(qrow + khalf * 16 + 8));
#pragma unroll
      for (int mt = 0; mt < 4; ++mt) {
        const __bf16* kcol = Ks + (mt * 16 + laneN) * APQ + ks * 32 + khalf * 16;
        v16bf b = cat16(lds_ld8(kcol), lds_ld8(kcol + 8));
        s[mt] = wmma_bf16(a, b, s[mt]);
      }
    }

    // ---- streaming softmax update (rows live in 16-lane halves) ----
#pragma unroll
    for (int r = 0; r < 8; ++r) {
      float mx = fmaxf(fmaxf(s[0][r], s[1][r]), fmaxf(s[2][r], s[3][r]));
#pragma unroll
      for (int off = 1; off < 16; off <<= 1) mx = fmaxf(mx, __shfl_xor(mx, off));
      float nm   = fmaxf(rmax[r], mx);
      float corr = __expf(rmax[r] - nm);
      rmax[r] = nm;
      float pv[4], psum = 0.0f;
#pragma unroll
      for (int mt = 0; mt < 4; ++mt) { pv[mt] = __expf(s[mt][r] - nm); psum += pv[mt]; }
#pragma unroll
      for (int off = 1; off < 16; off <<= 1) psum += __shfl_xor(psum, off);
      rsum[r] = rsum[r] * corr + psum;
#pragma unroll
      for (int et = 0; et < 4; ++et) acco[et][r] *= corr;
      int lr = wrow0 + r + 8 * khalf;          // this wave's own P rows
#pragma unroll
      for (int mt = 0; mt < 4; ++mt) Ps[lr * APQ + mt * 16 + laneN] = (__bf16)pv[mt];
    }
    // P is produced & consumed by the same wave; per-wave DS ops are in-order.

    // ---- O += P (16l x 64m) @ V^T (64m x 64e) ----
#pragma unroll
    for (int ks = 0; ks < 2; ++ks) {
      const __bf16* prow = Ps + (wrow0 + laneN) * APQ + ks * 32;
      v16bf a = cat16(lds_ld8(prow + khalf * 16), lds_ld8(prow + khalf * 16 + 8));
#pragma unroll
      for (int et = 0; et < 4; ++et) {
        const __bf16* vrow = Vs + (et * 16 + laneN) * APQ + ks * 32 + khalf * 16;
        v16bf b = cat16(lds_ld8(vrow), lds_ld8(vrow + 8));
        acco[et] = wmma_bf16(a, b, acco[et]);
      }
    }
    __syncthreads();                            // before K/V restage
  }

  // ---- normalize + store O pos-major bf16 [N][L][C] ----
#pragma unroll
  for (int r = 0; r < 8; ++r) {
    float inv = 1.0f / rsum[r];
    int l = l0 + wrow0 + r + 8 * khalf;
#pragma unroll
    for (int et = 0; et < 4; ++et) {
      Ob[((size_t)(n * L_DIM + l)) * C_DIM + h * E_DIM + et * 16 + laneN] =
          (__bf16)(acco[et][r] * inv);
    }
  }
}

// ---------------- host-side orchestration ----------------
extern "C" void kernel_launch(void* const* d_in, const int* in_sizes, int n_in,
                              void* d_out, int out_size, void* d_ws, size_t ws_size,
                              hipStream_t stream) {
  const float* x  = (const float*)d_in[0];
  const float* Wq = (const float*)d_in[1];
  const float* bq = (const float*)d_in[2];
  const float* Wk = (const float*)d_in[3];
  const float* bk = (const float*)d_in[4];
  const float* Wv = (const float*)d_in[5];
  const float* bv = (const float*)d_in[6];
  const float* Wo = (const float*)d_in[7];
  const float* bo = (const float*)d_in[8];
  const float* Wa = (const float*)d_in[9];
  const float* g1 = (const float*)d_in[10];
  const float* b1 = (const float*)d_in[11];
  const float* m1 = (const float*)d_in[12];
  const float* v1 = (const float*)d_in[13];
  const float* g2 = (const float*)d_in[14];
  const float* b2 = (const float*)d_in[15];
  const float* m2 = (const float*)d_in[16];
  const float* v2 = (const float*)d_in[17];

  // workspace layout (byte-based; all offsets 16B-aligned)
  char* wsb = (char*)d_ws;
  const size_t CL2 = (size_t)N_B * C_DIM * L2_DIM;   // 1M elements
  const size_t CL  = (size_t)N_B * C_DIM * L_DIM;    // 4M elements
  float*  bnS = (float*)wsb;                    wsb += 256 * sizeof(float);
  float*  bnT = (float*)wsb;                    wsb += 256 * sizeof(float);
  float*  xa0 = (float*)wsb;                    wsb += CL2 * sizeof(float);
  float*  xa2 = (float*)wsb;                    wsb += CL2 * sizeof(float);
  __bf16* Qbf = (__bf16*)wsb;                   wsb += CL  * sizeof(__bf16);
  __bf16* Kbf = (__bf16*)wsb;                   wsb += CL2 * sizeof(__bf16);
  __bf16* Vbf = (__bf16*)wsb;                   wsb += CL2 * sizeof(__bf16);
  __bf16* Obf = (__bf16*)wsb;                   wsb += CL  * sizeof(__bf16);
  float*  out = (float*)d_out;

  const float SM_SCALE = 0.125f;                // 1/sqrt(E)

  // 1) fold BN1∘BN2
  bn_coeffs_kernel<<<1, 256, 0, stream>>>(g1, b1, m1, v1, g2, b2, m2, v2, bnS, bnT);
  // 2) avg+max pool  x -> xa0 [N][C][L2] f32
  pool_kernel<<<(N_B * C_DIM * L2_DIM) / 256, 256, 0, stream>>>(x, xa0);
  // 3) q = (Wq x + bq) * 1/sqrt(E) -> Qbf pos-major bf16
  gemm256_kernel<false, true, false, false, true>
      <<<dim3(L_DIM / 64, 2, N_B), 256, 0, stream>>>(
          Wq, x, Qbf, nullptr, bq, SM_SCALE, L_DIM);
  // 4) xa2 = BN2(BN1(Wa xa0)) -> f32 [N][C][L2]
  gemm256_kernel<false, false, true, false, false>
      <<<dim3(L2_DIM / 64, 2, N_B), 256, 0, stream>>>(
          Wa, xa0, xa2, bnS, bnT, 1.0f, L2_DIM);
  // 5) k = Wk xa2 + bk -> Kbf pos-major bf16
  gemm256_kernel<false, true, false, false, true>
      <<<dim3(L2_DIM / 64, 2, N_B), 256, 0, stream>>>(
          Wk, xa2, Kbf, nullptr, bk, 1.0f, L2_DIM);
  // 6) v = Wv xa2 + bv -> Vbf channel-major bf16
  gemm256_kernel<false, false, false, false, true>
      <<<dim3(L2_DIM / 64, 2, N_B), 256, 0, stream>>>(
          Wv, xa2, Vbf, nullptr, bv, 1.0f, L2_DIM);
  // 7) flash attention -> Obf pos-major bf16
  attn_kernel<<<dim3(L_DIM / 64, HEADS, N_B), 128, 0, stream>>>(Qbf, Kbf, Vbf, Obf);
  // 8) out = Wo o + bo -> f32 [N][C][L]
  gemm256_kernel<true, false, false, true, false>
      <<<dim3(L_DIM / 64, 2, N_B), 256, 0, stream>>>(
          Wo, Obf, out, nullptr, bo, 1.0f, L_DIM);
}
